// CausalPrior_70128226009840
// MI455X (gfx1250) — matrix operand order repaired and checked by reading, not compiled, not run on hardware
//
#include <hip/hip_runtime.h>

typedef __attribute__((ext_vector_type(16))) _Float16 v16h;
typedef __attribute__((ext_vector_type(8)))  float    v8f;

#define B_SZ     32
#define N_NODES  4096
#define T_STEPS  16
#define N_COMM   32
#define DLAT     8
#define HLF      4
#define NBLK     3
#define HID      64
#define TILES    (B_SZ * N_NODES)       // 131072 tiles of 16 rows
#define WPB      4                       // waves per block
#define TPW      16                      // tiles per wave (amortize LDS staging)
#define NB_MAIN  (TILES / (WPB * TPW))   // 2048 blocks

// ---------------------------------------------------------------------------
// Wave-local LDS ordering: CDNA5 split counters + compiler fence.
// ---------------------------------------------------------------------------
__device__ __forceinline__ void wave_sync() {
  __builtin_amdgcn_wave_barrier();
  asm volatile("s_wait_dscnt 0x0" ::: "memory");
  __builtin_amdgcn_wave_barrier();
}

// ===========================================================================
// Kernel 1: per-community spectral filter G = Re(V diag(lam*mask) V^-1)
// Complex QR algorithm (Wilkinson shift, Givens), 1 thread per 8x8 matrix.
// ===========================================================================
struct cx { float r, i; };
__device__ __forceinline__ cx cadd(cx a, cx b){ return {a.r+b.r, a.i+b.i}; }
__device__ __forceinline__ cx csub(cx a, cx b){ return {a.r-b.r, a.i-b.i}; }
__device__ __forceinline__ cx cmul(cx a, cx b){ return {a.r*b.r - a.i*b.i, a.r*b.i + a.i*b.r}; }
__device__ __forceinline__ cx cconjf(cx a){ return {a.r, -a.i}; }
__device__ __forceinline__ float cabs2f(cx a){ return a.r*a.r + a.i*a.i; }
__device__ __forceinline__ float cmagf(cx a){ return sqrtf(cabs2f(a)); }
__device__ __forceinline__ cx cdivf(cx a, cx b){
  float d = cabs2f(b) + 1e-30f;
  return { (a.r*b.r + a.i*b.i)/d, (a.i*b.r - a.r*b.i)/d };
}
__device__ __forceinline__ cx csqrtf_(cx z){
  float m = cmagf(z);
  float re = sqrtf(fmaxf(0.f, 0.5f*(m + z.r)));
  float im = sqrtf(fmaxf(0.f, 0.5f*(m - z.r)));
  if (z.i < 0.f) im = -im;
  return {re, im};
}

__global__ void eig_graphs_kernel(const float* __restrict__ koop,
                                  float* __restrict__ graphs) {
  int c = blockIdx.x * blockDim.x + threadIdx.x;
  if (c >= N_COMM) return;
  const float* A = koop + c * DLAT * DLAT;
  cx H[8][8], Q[8][8];
  for (int i = 0; i < 8; ++i)
    for (int j = 0; j < 8; ++j) {
      H[i][j] = { A[i*8 + j], 0.f };
      Q[i][j] = { (i == j) ? 1.f : 0.f, 0.f };
    }
  // Shifted QR with deflation -> complex Schur form (H upper-tri), Q unitary
  for (int k = 7; k >= 1; --k) {
    for (int it = 0; it < 80; ++it) {
      float off = cmagf(H[k][k-1]);
      if (off <= 1e-7f * (cmagf(H[k-1][k-1]) + cmagf(H[k][k]) + 1e-20f)) break;
      cx a = H[k-1][k-1], b = H[k-1][k], cc = H[k][k-1], d = H[k][k];
      cx tr = cadd(a, d);
      cx det = csub(cmul(a, d), cmul(b, cc));
      cx disc = csqrtf_(csub(cmul(tr, tr), cx{4.f*det.r, 4.f*det.i}));
      cx l1 = { 0.5f*(tr.r + disc.r), 0.5f*(tr.i + disc.i) };
      cx l2 = { 0.5f*(tr.r - disc.r), 0.5f*(tr.i - disc.i) };
      cx mu = (cabs2f(csub(l1, d)) < cabs2f(csub(l2, d))) ? l1 : l2;
      for (int i = 0; i <= k; ++i) H[i][i] = csub(H[i][i], mu);
      float cs[8]; cx sn[8];
      for (int i = 0; i < k; ++i) {
        cx aa = H[i][i], bb = H[i+1][i];
        float r = sqrtf(cabs2f(aa) + cabs2f(bb));
        float ca; cx sa;
        if (r < 1e-30f) { ca = 1.f; sa = {0.f, 0.f}; }
        else {
          float ma = cmagf(aa);
          if (ma < 1e-30f) {
            float mb = cmagf(bb);
            ca = 0.f; sa = { bb.r/mb, -bb.i/mb };
          } else {
            ca = ma / r;
            cx ph = { aa.r/ma, aa.i/ma };
            sa = cmul(ph, cconjf(bb));
            sa.r /= r; sa.i /= r;
          }
        }
        cs[i] = ca; sn[i] = sa;
        for (int j = 0; j < 8; ++j) {
          cx h0 = H[i][j], h1 = H[i+1][j];
          H[i][j]   = cadd(cx{ca*h0.r, ca*h0.i}, cmul(sa, h1));
          H[i+1][j] = csub(cx{ca*h1.r, ca*h1.i}, cmul(cconjf(sa), h0));
        }
      }
      for (int i = 0; i < k; ++i) {
        float ca = cs[i]; cx sa = sn[i];
        for (int j = 0; j < 8; ++j) {
          cx h0 = H[j][i], h1 = H[j][i+1];
          H[j][i]   = cadd(cx{ca*h0.r, ca*h0.i}, cmul(cconjf(sa), h1));
          H[j][i+1] = csub(cx{ca*h1.r, ca*h1.i}, cmul(sa, h0));
          cx q0 = Q[j][i], q1 = Q[j][i+1];
          Q[j][i]   = cadd(cx{ca*q0.r, ca*q0.i}, cmul(cconjf(sa), q1));
          Q[j][i+1] = csub(cx{ca*q1.r, ca*q1.i}, cmul(sa, q0));
        }
      }
      for (int i = 0; i <= k; ++i) H[i][i] = cadd(H[i][i], mu);
    }
    H[k][k-1] = {0.f, 0.f};
  }
  // Eigenvectors: back-substitution on triangular H, then V = Q*y
  cx V[8][8];
  for (int kk = 0; kk < 8; ++kk) {
    cx y[8];
    for (int j = 0; j < 8; ++j) y[j] = {0.f, 0.f};
    y[kk] = {1.f, 0.f};
    for (int j = kk - 1; j >= 0; --j) {
      cx s = {0.f, 0.f};
      for (int i = j + 1; i <= kk; ++i) s = cadd(s, cmul(H[j][i], y[i]));
      cx den = csub(H[j][j], H[kk][kk]);
      if (cabs2f(den) < 1e-24f) den = {1e-12f, 1e-12f};
      y[j] = cdivf(cx{-s.r, -s.i}, den);
    }
    float nrm = 0.f;
    for (int i = 0; i < 8; ++i) {
      cx acc = {0.f, 0.f};
      for (int j = 0; j <= kk; ++j) acc = cadd(acc, cmul(Q[i][j], y[j]));
      V[i][kk] = acc; nrm += cabs2f(acc);
    }
    float inv = 1.f / sqrtf(nrm + 1e-30f);
    for (int i = 0; i < 8; ++i) { V[i][kk].r *= inv; V[i][kk].i *= inv; }
  }
  // V^-1 via Gauss-Jordan with partial pivoting
  cx M[8][16];
  for (int i = 0; i < 8; ++i)
    for (int j = 0; j < 8; ++j) {
      M[i][j] = V[i][j];
      M[i][8 + j] = { (i == j) ? 1.f : 0.f, 0.f };
    }
  for (int col = 0; col < 8; ++col) {
    int piv = col; float best = cabs2f(M[col][col]);
    for (int rr = col + 1; rr < 8; ++rr) {
      float v = cabs2f(M[rr][col]);
      if (v > best) { best = v; piv = rr; }
    }
    if (piv != col)
      for (int j = 0; j < 16; ++j) { cx t = M[col][j]; M[col][j] = M[piv][j]; M[piv][j] = t; }
    cx p = M[col][col];
    if (cabs2f(p) < 1e-30f) p = {1e-15f, 0.f};
    for (int j = 0; j < 16; ++j) M[col][j] = cdivf(M[col][j], p);
    for (int rr = 0; rr < 8; ++rr) {
      if (rr == col) continue;
      cx f = M[rr][col];
      for (int j = 0; j < 16; ++j) M[rr][j] = csub(M[rr][j], cmul(f, M[col][j]));
    }
  }
  // Mask eigenvalues |lam| in [0.9, 1.1] (fallback: keep all)
  cx lam[8]; bool msk[8]; bool any = false;
  for (int j = 0; j < 8; ++j) {
    lam[j] = cadd(H[j][j], cx{1e-10f, 1e-10f});
    float a = cmagf(lam[j]);
    msk[j] = (a <= 1.1f) && (a >= 0.9f);
    any |= msk[j];
  }
  for (int j = 0; j < 8; ++j) if (!any) msk[j] = true;
  float* Gout = graphs + c * 64;
  for (int i = 0; i < 8; ++i)
    for (int k2 = 0; k2 < 8; ++k2) {
      cx acc = {0.f, 0.f};
      for (int j = 0; j < 8; ++j) {
        if (!msk[j]) continue;
        acc = cadd(acc, cmul(cmul(V[i][j], lam[j]), M[j][8 + k2]));
      }
      Gout[i*8 + k2] = acc.r;
    }
}

// ===========================================================================
// Kernel 2: community argmax per node
// ===========================================================================
__global__ void classify_kernel(const float* __restrict__ comm,
                                int* __restrict__ cls) {
  int n = blockIdx.x * blockDim.x + threadIdx.x;
  if (n >= N_NODES) return;
  const float* row = comm + n * N_COMM;
  int best = 0; float bv = row[0];
  for (int c = 1; c < N_COMM; ++c) {
    float v = row[c];
    if (v > bv) { bv = v; best = c; }
  }
  cls[n] = best;
}

// ===========================================================================
// Kernel 3: transpose/pad/f16-convert weights (B-fragment friendly: [N][K])
//   w0t: [3][64][32]  (K 12->32 zero pad)
//   w1t: [3][64][64]
//   w2t: [3][16][64]  (N 8->16 zero pad)
//   b2p: [3][16]      (zero pad)
// ===========================================================================
__global__ void prep_weights_kernel(const float* __restrict__ W0,
                                    const float* __restrict__ W1,
                                    const float* __restrict__ W2,
                                    const float* __restrict__ b2,
                                    _Float16* __restrict__ w0t,
                                    _Float16* __restrict__ w1t,
                                    _Float16* __restrict__ w2t,
                                    float* __restrict__ b2p) {
  int tid = threadIdx.x;
  for (int idx = tid; idx < 3*64*32; idx += blockDim.x) {
    int i = idx / (64*32), n = (idx / 32) % 64, k = idx % 32;
    w0t[idx] = (k < 12) ? (_Float16)W0[(i*12 + k)*64 + n] : (_Float16)0.f;
  }
  for (int idx = tid; idx < 3*64*64; idx += blockDim.x) {
    int i = idx / 4096, n = (idx / 64) % 64, k = idx % 64;
    w1t[idx] = (_Float16)W1[(i*64 + k)*64 + n];
  }
  for (int idx = tid; idx < 3*16*64; idx += blockDim.x) {
    int i = idx / 1024, n = (idx / 64) % 16, k = idx % 64;
    w2t[idx] = (n < 8) ? (_Float16)W2[(i*64 + k)*8 + n] : (_Float16)0.f;
  }
  for (int idx = tid; idx < 3*16; idx += blockDim.x) {
    int i = idx / 16, n = idx % 16;
    b2p[idx] = (n < 8) ? b2[i*8 + n] : 0.f;
  }
}

// ===========================================================================
// Kernel 4: main coupling-flow kernel. One wave = TPW tiles of 16 rows each.
// ===========================================================================
struct __align__(16) SMem {
  _Float16 w0t[3][64][32];   // 12288 B (contiguous w0t..w2t = 43008 B in ws)
  _Float16 w1t[3][64][64];   // 24576 B
  _Float16 w2t[3][16][64];   //  6144 B
  float b0s[3][64];          //   768 B
  float b1s[3][64];          //   768 B
  float b2s[3][16];          //   192 B
  _Float16 inbuf[WPB][16][32]; // 4096 B  layer-0 inputs / x staging (A frags)
  _Float16 hbuf [WPB][16][64]; // 8192 B  hidden activations
  float    xbuf [WPB][16][8];  // 2048 B  flow state (f32)
  float    obuf [WPB][16][8];  // 2048 B  last-layer outputs (s|t)
};                             // total = 61120 B

union AFrag { v16h v; uint4 q[2]; };

__global__ void __launch_bounds__(WPB * 32)
coupling_flow_kernel(const float* __restrict__ latent,
                     const float* __restrict__ graphs,
                     const int* __restrict__ cls,
                     const _Float16* __restrict__ wpack,   // w0t|w1t|w2t in ws
                     const float* __restrict__ b0,
                     const float* __restrict__ b1,
                     const float* __restrict__ b2p,
                     float* __restrict__ out) {
  __shared__ SMem sm;
  const int tid  = threadIdx.x;
  const int wave = tid >> 5;
  const int lane = tid & 31;

  // ---- stage weights into LDS (once per block, reused by TPW*WPB tiles) ----
  {
    uint4* dst = (uint4*)&sm.w0t[0][0][0];
    const uint4* src = (const uint4*)wpack;
    for (int i = tid; i < 43008/16; i += WPB*32) dst[i] = src[i];
    for (int i = tid; i < 3*64; i += WPB*32) {
      ((float*)sm.b0s)[i] = b0[i];
      ((float*)sm.b1s)[i] = b1[i];
    }
    for (int i = tid; i < 3*16; i += WPB*32) ((float*)sm.b2s)[i] = b2p[i];
  }
  __syncthreads();

  const int N  = lane & 15;      // fragment column
  const int hi = lane >> 4;      // half-wave select
  const int kb = hi * 8;         // A-fragment K base
  const int M  = N;              // A-fragment row

  _Float16 (*ib)[32] = sm.inbuf[wave];
  _Float16 (*hb)[64] = sm.hbuf[wave];
  float    (*xb)[8]  = sm.xbuf[wave];
  float    (*ob)[8]  = sm.obuf[wave];

  const v8f vzero = {0.f,0.f,0.f,0.f,0.f,0.f,0.f,0.f};

  // Zero inbuf once: K-columns 12..31 stay zero forever; 0..11 are fully
  // rewritten each tile (stale cond in 8..11 hits zero rows of padded B_G).
  {
    const uint4 z4 = {0u,0u,0u,0u};
    ((uint4*)&ib[0][0])[lane]      = z4;
    ((uint4*)&ib[0][0])[lane + 32] = z4;
  }

  // Consecutive tiles across waves each step -> contiguous latent traffic.
  const int tile0 = blockIdx.x * (WPB * TPW) + wave;

  #pragma unroll 1
  for (int tt = 0; tt < TPW; ++tt) {
    const int tile  = tile0 + tt * WPB;
    const int node  = tile & (N_NODES - 1);
    const long row0 = (long)tile * 16;

    // prefetch next tile's latent rows (global_prefetch_b8)
    if (tt + 1 < TPW)
      __builtin_prefetch(latent + ((long)(tile + WPB) * 16) * 8 + lane * 4, 0, 0);

    // ---- load x tile (16 rows x 8 f32) ----
    {
      const float4 xv = ((const float4*)(latent + row0 * 8))[lane];
      ((float4*)&xb[0][0])[lane] = xv;
    }
    wave_sync();
    if (lane < 16) {
      #pragma unroll
      for (int d = 0; d < 8; ++d) ib[lane][d] = (_Float16)xb[lane][d];
    }
    wave_sync();

    // ---- B fragment for G (8x8 padded to 32x16) ----
    v16h bgv;
    #pragma unroll
    for (int e = 0; e < 16; ++e) bgv[e] = (_Float16)0.f;
    {
      const int cn = cls[node];
      const float* Gp = graphs + cn * 64;
      if (hi == 0 && N < 8) {
        #pragma unroll
        for (int e = 0; e < 8; ++e) bgv[e] = (_Float16)Gp[e*8 + N];  // B[K=e][N]
      }
    }

    // ---- cond = x @ G  (one WMMA) ----
    AFrag ax;
    ax.q[0] = *(const uint4*)&ib[M][kb];
    ax.q[1] = *(const uint4*)&ib[M][16 + kb];
    v8f cond = __builtin_amdgcn_wmma_f32_16x16x32_f16(
        false, ax.v, false, bgv, (short)0, vzero, false, false);
    if (N < 8) {
      #pragma unroll
      for (int r = 0; r < 8; ++r)
        ib[r + hi*8][4 + N] = (_Float16)cond[r];   // cond -> K slots 4..11
    }
    wave_sync();

    float ld = 0.f;   // logdet accumulator (valid on lanes 0..15)

    #pragma unroll
    for (int blk = 0; blk < NBLK; ++blk) {
      const int d0 = (blk & 1) ? 4 : 0;   // x1 source half
      if (lane < 16) {
        #pragma unroll
        for (int d = 0; d < 4; ++d) ib[lane][d] = (_Float16)xb[lane][d0 + d];
      }
      wave_sync();

      // ---- layer 0: [x1|cond] (K=32 padded) @ W0 -> h (16x64) ----
      AFrag ain;
      ain.q[0] = *(const uint4*)&ib[M][kb];
      ain.q[1] = *(const uint4*)&ib[M][16 + kb];
      #pragma unroll
      for (int nt = 0; nt < 4; ++nt) {
        const int nc = nt*16 + N;
        AFrag bw;
        const _Float16* wr = &sm.w0t[blk][nc][0];
        bw.q[0] = *(const uint4*)&wr[hi*8];
        bw.q[1] = *(const uint4*)&wr[16 + hi*8];
        v8f acc = __builtin_amdgcn_wmma_f32_16x16x32_f16(
            false, ain.v, false, bw.v, (short)0, vzero, false, false);
        const float bias = sm.b0s[blk][nc];
        #pragma unroll
        for (int r = 0; r < 8; ++r)
          hb[r + hi*8][nc] = (_Float16)fmaxf(acc[r] + bias, 0.f);
      }
      wave_sync();

      // ---- layer 1: h @ W1 (K=64) -> h ----
      AFrag ah0, ah1;
      ah0.q[0] = *(const uint4*)&hb[M][kb];
      ah0.q[1] = *(const uint4*)&hb[M][16 + kb];
      ah1.q[0] = *(const uint4*)&hb[M][32 + kb];
      ah1.q[1] = *(const uint4*)&hb[M][48 + kb];
      wave_sync();
      #pragma unroll
      for (int nt = 0; nt < 4; ++nt) {
        const int nc = nt*16 + N;
        const _Float16* wr = &sm.w1t[blk][nc][0];
        AFrag bw;
        bw.q[0] = *(const uint4*)&wr[hi*8];
        bw.q[1] = *(const uint4*)&wr[16 + hi*8];
        v8f acc = __builtin_amdgcn_wmma_f32_16x16x32_f16(
            false, ah0.v, false, bw.v, (short)0, vzero, false, false);
        bw.q[0] = *(const uint4*)&wr[32 + hi*8];
        bw.q[1] = *(const uint4*)&wr[32 + 16 + hi*8];
        acc = __builtin_amdgcn_wmma_f32_16x16x32_f16(
            false, ah1.v, false, bw.v, (short)0, acc, false, false);
        const float bias = sm.b1s[blk][nc];
        #pragma unroll
        for (int r = 0; r < 8; ++r)
          hb[r + hi*8][nc] = (_Float16)fmaxf(acc[r] + bias, 0.f);
      }
      wave_sync();

      // ---- layer 2: h @ W2 (K=64, N=16 padded) -> o = [s|t] ----
      AFrag a20, a21;
      a20.q[0] = *(const uint4*)&hb[M][kb];
      a20.q[1] = *(const uint4*)&hb[M][16 + kb];
      a21.q[0] = *(const uint4*)&hb[M][32 + kb];
      a21.q[1] = *(const uint4*)&hb[M][48 + kb];
      {
        const _Float16* wr = &sm.w2t[blk][N][0];
        AFrag bw;
        bw.q[0] = *(const uint4*)&wr[hi*8];
        bw.q[1] = *(const uint4*)&wr[16 + hi*8];
        v8f o = __builtin_amdgcn_wmma_f32_16x16x32_f16(
            false, a20.v, false, bw.v, (short)0, vzero, false, false);
        bw.q[0] = *(const uint4*)&wr[32 + hi*8];
        bw.q[1] = *(const uint4*)&wr[32 + 16 + hi*8];
        o = __builtin_amdgcn_wmma_f32_16x16x32_f16(
            false, a21.v, false, bw.v, (short)0, o, false, false);
        const float bias2 = sm.b2s[blk][N];
        if (N < 8) {
          #pragma unroll
          for (int r = 0; r < 8; ++r) ob[r + hi*8][N] = o[r] + bias2;
        }
      }
      wave_sync();

      // ---- affine update: x2 = x2*exp(s) + t ; logdet += sum(s) ----
      if (lane < 16) {
        const int i2 = (blk & 1) ? 0 : 4;   // updated half
        float ss = 0.f;
        #pragma unroll
        for (int d = 0; d < 4; ++d) {
          const float s = tanhf(ob[lane][d]);
          const float t = ob[lane][4 + d];
          xb[lane][i2 + d] = xb[lane][i2 + d] * expf(s) + t;
          ss += s;
        }
        ld += ss;
      }
      wave_sync();
    }

    // ---- log_pz = -0.5*|z|^2 - 0.5*D*log(2*pi) + logdet ----
    if (lane < 16) {
      float z2 = 0.f;
      #pragma unroll
      for (int d = 0; d < 8; ++d) z2 += xb[lane][d] * xb[lane][d];
      out[row0 + lane] = -0.5f * z2 - 7.3515082656373816f + ld;
    }
    wave_sync();
  }
}

// ===========================================================================
// Launch
// ===========================================================================
extern "C" void kernel_launch(void* const* d_in, const int* in_sizes, int n_in,
                              void* d_out, int out_size, void* d_ws, size_t ws_size,
                              hipStream_t stream) {
  const float* latent = (const float*)d_in[0];
  const float* koop   = (const float*)d_in[1];
  const float* comm   = (const float*)d_in[2];
  const float* W0     = (const float*)d_in[3];
  const float* b0     = (const float*)d_in[4];
  const float* W1     = (const float*)d_in[5];
  const float* b1     = (const float*)d_in[6];
  const float* W2     = (const float*)d_in[7];
  const float* b2     = (const float*)d_in[8];
  float* out = (float*)d_out;
  char* ws = (char*)d_ws;

  // workspace layout (bytes)
  float*    graphs = (float*)(ws);                 // 32*64*4      = 8192
  int*      cls    = (int*)(ws + 8192);            // 4096*4       = 16384
  _Float16* w0t    = (_Float16*)(ws + 24576);      // 3*64*32*2    = 12288
  _Float16* w1t    = (_Float16*)(ws + 36864);      // 3*64*64*2    = 24576
  _Float16* w2t    = (_Float16*)(ws + 61440);      // 3*16*64*2    = 6144
  float*    b2p    = (float*)(ws + 67584);         // 3*16*4       = 192

  eig_graphs_kernel<<<1, 32, 0, stream>>>(koop, graphs);
  classify_kernel<<<N_NODES/256, 256, 0, stream>>>(comm, cls);
  prep_weights_kernel<<<1, 256, 0, stream>>>(W0, W1, W2, b2, w0t, w1t, w2t, b2p);
  coupling_flow_kernel<<<NB_MAIN, WPB*32, 0, stream>>>(
      latent, graphs, cls, w0t, b0, b1, b2p, out);
}
